// GatedSdpaAttention_17497696764015
// MI455X (gfx1250) — matrix operand order repaired and checked by reading, not compiled, not run on hardware
//
#include <hip/hip_runtime.h>
#include <hip/hip_bf16.h>

typedef __attribute__((ext_vector_type(16))) _Float16 v16h;
typedef __attribute__((ext_vector_type(8)))  _Float16 v8h;
typedef __attribute__((ext_vector_type(4)))  _Float16 v4h;
typedef __attribute__((ext_vector_type(8)))  float    v8f;

#define B_  4
#define L_  2048
#define E_  1024
#define H_  16
#define D_  64
#define M_  (B_ * L_)          // 8192
#define NQKV 4096              // 2*H*D + H*D + H*D

// ---------- helpers ----------

__device__ __forceinline__ v16h ld2x8(const _Float16* p0, const _Float16* p1) {
    union { v16h v; v8h h[2]; } u;
    u.h[0] = *(const v8h*)p0;   // 16-byte aligned chunk
    u.h[1] = *(const v8h*)p1;   // 16-byte aligned chunk
    return u.v;
}

__device__ __forceinline__ v8f wmma16(v16h a, v16h b, v8f c) {
    // D(16x16 f32) = A(16x32 f16) * B(32x16 f16) + C
    return __builtin_amdgcn_wmma_f32_16x16x32_f16(
        /*neg_a=*/false, a, /*neg_b=*/false, b,
        /*c_mod=*/(short)0, c, /*reuse_a=*/false, /*reuse_b=*/false);
}

// ---------- kernel 1: fused QKV projection  qkv[M][4096](f16) = h[M][1024] * [wq|wk|wv] ----------
// Block tile: BM=128, BN=64, BK=32.  8 waves, each wave owns 16 rows x 64 cols (4 accumulators).

__global__ void __launch_bounds__(256)
qkv_gemm_kernel(const float* __restrict__ h,
                const float* __restrict__ wq,
                const float* __restrict__ wk,
                const float* __restrict__ wv,
                _Float16* __restrict__ qkv)
{
    __shared__ _Float16 As[128][40];   // [m][k], 80B row stride (16B multiple)
    __shared__ _Float16 BsT[64][40];   // [n][k]

    const int tid  = threadIdx.x;
    const int m0   = blockIdx.y * 128;
    const int n0   = blockIdx.x * 64;
    const int wave = tid >> 5, lane = tid & 31;
    const int row16 = lane & 15, half = lane >> 4;

    // uniform weight-region select for this block (n0 is 64-aligned; 2048/3072 are 64-multiples)
    const float* wbase; int wstride, wcol0;
    if (n0 < 2048)      { wbase = wq; wstride = 2048; wcol0 = n0; }
    else if (n0 < 3072) { wbase = wk; wstride = 1024; wcol0 = n0 - 2048; }
    else                { wbase = wv; wstride = 1024; wcol0 = n0 - 3072; }

    v8f acc0 = {}, acc1 = {}, acc2 = {}, acc3 = {};

    for (int k0 = 0; k0 < E_; k0 += 32) {
        // --- stage A: 128x32 f32 -> f16, vectorized float4 loads + ds_store_b64 ---
#pragma unroll
        for (int it = 0; it < 4; ++it) {
            const int row = (tid >> 3) + it * 32;
            const int c4  = (tid & 7) * 4;
            const float4 f = *(const float4*)&h[(size_t)(m0 + row) * E_ + k0 + c4];
            v4h p = { (_Float16)f.x, (_Float16)f.y, (_Float16)f.z, (_Float16)f.w };
            *(v4h*)&As[row][c4] = p;
        }
        // --- stage B^T: 32x64 f32 -> f16 (coalesced float4 along n, transposed scalar stores) ---
#pragma unroll
        for (int it = 0; it < 2; ++it) {
            const int kk = (tid >> 4) + it * 16;
            const int nn = (tid & 15) * 4;
            const float4 f = *(const float4*)&wbase[(size_t)(k0 + kk) * wstride + wcol0 + nn];
            BsT[nn + 0][kk] = (_Float16)f.x;
            BsT[nn + 1][kk] = (_Float16)f.y;
            BsT[nn + 2][kk] = (_Float16)f.z;
            BsT[nn + 3][kk] = (_Float16)f.w;
        }
        __syncthreads();

        const v16h a = ld2x8(&As[wave * 16 + row16][half * 16], &As[wave * 16 + row16][half * 16 + 8]);
        const v16h b0 = ld2x8(&BsT[ 0 + row16][half * 16], &BsT[ 0 + row16][half * 16 + 8]);
        const v16h b1 = ld2x8(&BsT[16 + row16][half * 16], &BsT[16 + row16][half * 16 + 8]);
        const v16h b2 = ld2x8(&BsT[32 + row16][half * 16], &BsT[32 + row16][half * 16 + 8]);
        const v16h b3 = ld2x8(&BsT[48 + row16][half * 16], &BsT[48 + row16][half * 16 + 8]);
        acc0 = wmma16(a, b0, acc0);
        acc1 = wmma16(a, b1, acc1);
        acc2 = wmma16(a, b2, acc2);
        acc3 = wmma16(a, b3, acc3);
        __syncthreads();
    }

#pragma unroll
    for (int r = 0; r < 8; ++r) {
        const size_t gm = (size_t)(m0 + wave * 16 + half * 8 + r) * NQKV + n0;
        qkv[gm +  0 + row16] = (_Float16)acc0[r];
        qkv[gm + 16 + row16] = (_Float16)acc1[r];
        qkv[gm + 32 + row16] = (_Float16)acc2[r];
        qkv[gm + 48 + row16] = (_Float16)acc3[r];
    }
}

// ---------- kernel 2: RMSNorm + partial RoPE; layout change for attention ----------
// Q, K -> [B][H][L][D] f16 ;  V -> transposed [B][H][D][L] f16

__global__ void __launch_bounds__(256)
prep_kernel(const _Float16* __restrict__ qkv,
            const float* __restrict__ cosb,
            const float* __restrict__ sinb,
            const float* __restrict__ qw,
            const float* __restrict__ kw,
            _Float16* __restrict__ Qo,
            _Float16* __restrict__ Ko,
            _Float16* __restrict__ Vt)
{
    const int wid  = blockIdx.x * 8 + (threadIdx.x >> 5);
    const int lane = threadIdx.x & 31;
    const int hh = wid & 15;
    const int l  = (wid >> 4) & (L_ - 1);
    const int b  = wid >> 15;

    const size_t rowbase = (size_t)(b * L_ + l) * NQKV;
    const int d0 = lane, d1 = lane + 32;

    const size_t cb = (size_t)(b * L_ + l) * D_;
    const float c0 = cosb[cb + d0], c1 = cosb[cb + d1];
    const float s0 = sinb[cb + d0], s1 = sinb[cb + d1];
    const bool  rot = lane < 24;                 // (d % 32) < 24 for both halves
    const float cp0 = rot ? c0 : 1.f, cp1 = rot ? c1 : 1.f;
    const float sp0 = rot ? s0 : 0.f, sp1 = rot ? s1 : 0.f;

    // ---- Q: rmsnorm + rope ----
    {
        float q0 = (float)qkv[rowbase + hh * 128 + d0];
        float q1 = (float)qkv[rowbase + hh * 128 + d1];
        float ss = q0 * q0 + q1 * q1;
        for (int off = 16; off > 0; off >>= 1) ss += __shfl_xor(ss, off, 32);
        float inv = rsqrtf(ss * (1.0f / 64.0f) + 1e-6f);
        q0 = q0 * inv * qw[d0];
        q1 = q1 * inv * qw[d1];
        float r0 = q0 * cp0 - q1 * sp0;          // rotate_half: rh[d0] = -q[d0+32]
        float r1 = q1 * cp1 + q0 * sp1;          //              rh[d1] =  q[d1-32]
        const size_t qb = ((size_t)(b * H_ + hh) * L_ + l) * D_;
        Qo[qb + d0] = (_Float16)r0;
        Qo[qb + d1] = (_Float16)r1;
    }
    // ---- K: rmsnorm + rope ----
    {
        float k0 = (float)qkv[rowbase + 2048 + hh * 64 + d0];
        float k1 = (float)qkv[rowbase + 2048 + hh * 64 + d1];
        float ss = k0 * k0 + k1 * k1;
        for (int off = 16; off > 0; off >>= 1) ss += __shfl_xor(ss, off, 32);
        float inv = rsqrtf(ss * (1.0f / 64.0f) + 1e-6f);
        k0 = k0 * inv * kw[d0];
        k1 = k1 * inv * kw[d1];
        float r0 = k0 * cp0 - k1 * sp0;
        float r1 = k1 * cp1 + k0 * sp1;
        const size_t kb = ((size_t)(b * H_ + hh) * L_ + l) * D_;
        Ko[kb + d0] = (_Float16)r0;
        Ko[kb + d1] = (_Float16)r1;
    }
    // ---- V: transpose to [B][H][D][L] ----
    {
        _Float16 v0 = qkv[rowbase + 3072 + hh * 64 + d0];
        _Float16 v1 = qkv[rowbase + 3072 + hh * 64 + d1];
        const size_t vb = (size_t)(b * H_ + hh) * D_;
        Vt[(vb + d0) * L_ + l] = v0;
        Vt[(vb + d1) * L_ + l] = v1;
    }
}

// ---------- kernel 3: flash attention + gate.  One wave per (b, h, 16-row q tile). ----------

__global__ void __launch_bounds__(256)
attn_kernel(const _Float16* __restrict__ Q,
            const _Float16* __restrict__ K,
            const _Float16* __restrict__ Vt,
            const float* __restrict__ mask,
            const _Float16* __restrict__ qkv,     // gate source
            _Float16* __restrict__ O)
{
    __shared__ _Float16 P[8][16][40];             // per-wave P tile (C-layout -> A-layout staging)

    const int wave = threadIdx.x >> 5, lane = threadIdx.x & 31;
    const int row16 = lane & 15, half = lane >> 4;
    const int wid = blockIdx.x * 8 + wave;
    const int qt = wid & 127;                     // L/16
    const int hh = (wid >> 7) & 15;
    const int b  = wid >> 11;

    const int qrow = qt * 16 + row16;
    const _Float16* qb = Q + ((size_t)(b * H_ + hh) * L_ + qrow) * D_;
    const v16h qa0 = ld2x8(qb + half * 16,      qb + half * 16 + 8);   // d 0..31
    const v16h qa1 = ld2x8(qb + 32 + half * 16, qb + 32 + half * 16 + 8);   // d 32..63

    v8f o0 = {}, o1 = {}, o2 = {}, o3 = {};
    float mrow[8], lrow[8];
#pragma unroll
    for (int r = 0; r < 8; ++r) { mrow[r] = -1e30f; lrow[r] = 0.f; }

    const float scale = 0.125f;                   // 1/sqrt(D)
    const _Float16* kbase = K + (size_t)(b * H_ + hh) * L_ * D_;
    const _Float16* vbase = Vt + (size_t)(b * H_ + hh) * D_ * L_;
    const size_t mbase = ((size_t)b * L_ + qt * 16 + half * 8) * L_;

    for (int kb = 0; kb < L_; kb += 32) {
        if (kb + 32 < L_)
            __builtin_prefetch(mask + mbase + kb + 32 + row16, 0, 0);

        // ---- scores: two 16x16 tiles over this 32-key block ----
        v8f s0 = {}, s1 = {};
        {
            const _Float16* kp0 = kbase + (size_t)(kb + row16) * D_;
            v16h f0 = ld2x8(kp0 + half * 16,      kp0 + half * 16 + 8);       // d 0..31
            v16h f1 = ld2x8(kp0 + 32 + half * 16, kp0 + 32 + half * 16 + 8);  // d 32..63
            s0 = wmma16(qa0, f0, s0);
            s0 = wmma16(qa1, f1, s0);
            const _Float16* kp1 = kp0 + 16 * D_;
            v16h g0 = ld2x8(kp1 + half * 16,      kp1 + half * 16 + 8);
            v16h g1 = ld2x8(kp1 + 32 + half * 16, kp1 + 32 + half * 16 + 8);
            s1 = wmma16(qa0, g0, s1);
            s1 = wmma16(qa1, g1, s1);
        }

        // ---- mask + online softmax (row stats over 16-lane groups) ----
        float p0a[8], p1a[8];
#pragma unroll
        for (int r = 0; r < 8; ++r) {
            const size_t mb = mbase + (size_t)r * L_ + kb;
            float v0 = s0[r] * scale + mask[mb + row16];
            float v1 = s1[r] * scale + mask[mb + 16 + row16];
            float x = fmaxf(v0, v1);
            for (int off = 8; off > 0; off >>= 1) x = fmaxf(x, __shfl_xor(x, off, 16));
            const float mn = fmaxf(mrow[r], x);
            const float sc = __expf(mrow[r] - mn);
            mrow[r] = mn;
            const float e0 = __expf(v0 - mn), e1 = __expf(v1 - mn);
            float ps = e0 + e1;
            for (int off = 8; off > 0; off >>= 1) ps += __shfl_xor(ps, off, 16);
            lrow[r] = lrow[r] * sc + ps;
            o0[r] *= sc; o1[r] *= sc; o2[r] *= sc; o3[r] *= sc;
            p0a[r] = e0; p1a[r] = e1;
        }

        // ---- P (C-layout) -> LDS -> A-layout fragment (wave-private, in-order LDS pipe) ----
#pragma unroll
        for (int r = 0; r < 8; ++r) {
            P[wave][half * 8 + r][row16]      = (_Float16)p0a[r];
            P[wave][half * 8 + r][16 + row16] = (_Float16)p1a[r];
        }
        const v16h pa = ld2x8(&P[wave][row16][half * 16], &P[wave][row16][half * 16 + 8]);

        // ---- PV: V^T rows give contiguous B-fragments ----
        {
            const _Float16* v0p = vbase + (size_t)(0 * 16 + row16) * L_ + kb + half * 16;
            const _Float16* v1p = vbase + (size_t)(1 * 16 + row16) * L_ + kb + half * 16;
            const _Float16* v2p = vbase + (size_t)(2 * 16 + row16) * L_ + kb + half * 16;
            const _Float16* v3p = vbase + (size_t)(3 * 16 + row16) * L_ + kb + half * 16;
            o0 = wmma16(pa, ld2x8(v0p, v0p + 8), o0);
            o1 = wmma16(pa, ld2x8(v1p, v1p + 8), o1);
            o2 = wmma16(pa, ld2x8(v2p, v2p + 8), o2);
            o3 = wmma16(pa, ld2x8(v3p, v3p + 8), o3);
        }
    }

    // ---- normalize, gate, store O[b][l][h*64+d] (f16 A-matrix for out-proj) ----
#pragma unroll
    for (int r = 0; r < 8; ++r) {
        const float invl = 1.0f / lrow[r];
        const int mrw = qt * 16 + half * 8 + r;
        const size_t grow = (size_t)(b * L_ + mrw);
#pragma unroll
        for (int g = 0; g < 4; ++g) {
            const int dcol = g * 16 + row16;
            const float gv = (float)qkv[grow * NQKV + hh * 128 + 64 + dcol];
            const float sig = 1.0f / (1.0f + __expf(-gv));
            float val;
            if      (g == 0) val = o0[r];
            else if (g == 1) val = o1[r];
            else if (g == 2) val = o2[r];
            else             val = o3[r];
            O[grow * (H_ * D_) + hh * 64 + dcol] = (_Float16)(val * invl * sig);
        }
    }
}

// ---------- kernel 4: output projection  out[M][1024](f32) = O[M][1024](f16) * wo ----------
// BM=128, BN=64, BK=32.  A tile staged with gfx1250 async LDS DMA (ASYNCcnt tracked).

__global__ void __launch_bounds__(256)
out_gemm_kernel(const _Float16* __restrict__ O,
                const float* __restrict__ wo,
                float* __restrict__ out)
{
    __shared__ _Float16 As[128][40];
    __shared__ _Float16 BsT[64][40];

    const int tid  = threadIdx.x;
    const int m0   = blockIdx.y * 128;
    const int n0   = blockIdx.x * 64;
    const int wave = tid >> 5, lane = tid & 31;
    const int row16 = lane & 15, half = lane >> 4;

    v8f acc0 = {}, acc1 = {}, acc2 = {}, acc3 = {};

    for (int k0 = 0; k0 < H_ * D_; k0 += 32) {
        // --- stage A: 128x32 f16 via async global->LDS DMA (16B per lane) ---
#pragma unroll
        for (int it = 0; it < 2; ++it) {
            const int row = (tid >> 2) + it * 64;
            const int c8  = (tid & 3) * 8;
            const _Float16* gp = &O[(size_t)(m0 + row) * (H_ * D_) + k0 + c8];
            const unsigned ldsoff = (unsigned)(size_t)&As[row][c8];
            asm volatile("global_load_async_to_lds_b128 %0, %1, off"
                         :: "v"(ldsoff), "v"(gp) : "memory");
        }
        // --- stage B^T: 32x64 f32 -> f16 ---
#pragma unroll
        for (int it = 0; it < 2; ++it) {
            const int kk = (tid >> 4) + it * 16;
            const int nn = (tid & 15) * 4;
            const float4 f = *(const float4*)&wo[(size_t)(k0 + kk) * E_ + n0 + nn];
            BsT[nn + 0][kk] = (_Float16)f.x;
            BsT[nn + 1][kk] = (_Float16)f.y;
            BsT[nn + 2][kk] = (_Float16)f.z;
            BsT[nn + 3][kk] = (_Float16)f.w;
        }
        asm volatile("s_wait_asynccnt 0" ::: "memory");
        __syncthreads();

        const v16h a = ld2x8(&As[wave * 16 + row16][half * 16], &As[wave * 16 + row16][half * 16 + 8]);
        const v16h b0 = ld2x8(&BsT[ 0 + row16][half * 16], &BsT[ 0 + row16][half * 16 + 8]);
        const v16h b1 = ld2x8(&BsT[16 + row16][half * 16], &BsT[16 + row16][half * 16 + 8]);
        const v16h b2 = ld2x8(&BsT[32 + row16][half * 16], &BsT[32 + row16][half * 16 + 8]);
        const v16h b3 = ld2x8(&BsT[48 + row16][half * 16], &BsT[48 + row16][half * 16 + 8]);
        acc0 = wmma16(a, b0, acc0);
        acc1 = wmma16(a, b1, acc1);
        acc2 = wmma16(a, b2, acc2);
        acc3 = wmma16(a, b3, acc3);
        __syncthreads();
    }

#pragma unroll
    for (int r = 0; r < 8; ++r) {
        const size_t gm = (size_t)(m0 + wave * 16 + half * 8 + r) * E_ + n0;
        out[gm +  0 + row16] = acc0[r];
        out[gm + 16 + row16] = acc1[r];
        out[gm + 32 + row16] = acc2[r];
        out[gm + 48 + row16] = acc3[r];
    }
}

// ---------- launch ----------

extern "C" void kernel_launch(void* const* d_in, const int* in_sizes, int n_in,
                              void* d_out, int out_size, void* d_ws, size_t ws_size,
                              hipStream_t stream) {
    const float* h    = (const float*)d_in[0];
    const float* cosb = (const float*)d_in[1];
    const float* sinb = (const float*)d_in[2];
    const float* mask = (const float*)d_in[3];
    const float* wq   = (const float*)d_in[4];
    const float* wk   = (const float*)d_in[5];
    const float* wv   = (const float*)d_in[6];
    const float* wo   = (const float*)d_in[7];
    const float* qw   = (const float*)d_in[8];
    const float* kw   = (const float*)d_in[9];
    float* out = (float*)d_out;

    char* ws = (char*)d_ws;
    const size_t SZ_QKV = (size_t)M_ * NQKV * 2;         // 64 MB  f16
    const size_t SZ_QKH = (size_t)B_ * H_ * L_ * D_ * 2; // 16 MB  f16 each
    _Float16* qkv = (_Float16*)(ws);
    _Float16* Qf  = (_Float16*)(ws + SZ_QKV);
    _Float16* Kf  = (_Float16*)(ws + SZ_QKV + SZ_QKH);
    _Float16* Vt  = (_Float16*)(ws + SZ_QKV + 2 * SZ_QKH);
    _Float16* Of  = (_Float16*)(ws + SZ_QKV + 3 * SZ_QKH);

    dim3 blk(256);
    qkv_gemm_kernel<<<dim3(NQKV / 64, M_ / 128), blk, 0, stream>>>(h, wq, wk, wv, qkv);
    prep_kernel<<<dim3((B_ * L_ * H_) / 8), blk, 0, stream>>>(qkv, cosb, sinb, qw, kw, Qf, Kf, Vt);
    attn_kernel<<<dim3((B_ * H_ * (L_ / 16)) / 8), blk, 0, stream>>>(Qf, Kf, Vt, mask, qkv, Of);
    out_gemm_kernel<<<dim3(E_ / 64, M_ / 128), blk, 0, stream>>>(Of, wo, out);
}
